// SelectiveKVCache_84963043049699
// MI455X (gfx1250) — matrix operand order repaired and checked by reading, not compiled, not run on hardware
//
#include <hip/hip_runtime.h>

// Problem constants (from reference setup_inputs)
#define BD   2
#define HD   32
#define SD   8192
#define DD   128
#define BH   (BD * HD)          // 64
#define FEAT 256                // 2*D
#define HID  256
#define KSEL 2048
#define FPAD 272                // 256 + 16 bf16 pad -> 544B row stride (32B aligned)

typedef __attribute__((ext_vector_type(16))) __bf16 v16bf;
typedef __attribute__((ext_vector_type(8)))  float  v8f;

// ---------------------------------------------------------------------------
// Kernel 1: importance scores via bf16 WMMA MLP
//   scores[t] = sum_c relu( (concat(k,v)[t] . W1[:,c]) + b1[c] ) * W2[c]
//              + b2 + recency(t % S)
// One wave per 16-token tile; 8 waves (256 threads) per block.
// W1 staged once per block into LDS as bf16, transposed [c][f] with padding.
// ---------------------------------------------------------------------------
__global__ __launch_bounds__(256)
void score_kernel(const float* __restrict__ k, const float* __restrict__ v,
                  const float* __restrict__ W1, const float* __restrict__ b1,
                  const float* __restrict__ W2, const float* __restrict__ b2,
                  float* __restrict__ scores)
{
    __shared__ __align__(64) __bf16 w1t[HID * FPAD];   // [c][f], ~136KB
    __shared__ float b1s[HID];
    __shared__ float w2s[HID];

    const int tid = threadIdx.x;

    // Stage W1 (global row-major [f][c]) -> LDS transposed bf16 [c][f]
    for (int e = tid; e < FEAT * HID; e += 256) {
        const int f = e >> 8;
        const int c = e & 255;
        w1t[c * FPAD + f] = (__bf16)W1[e];
    }
    for (int c = tid; c < HID; c += 256) { b1s[c] = b1[c]; w2s[c] = W2[c]; }
    __syncthreads();

    const int wave = tid >> 5;
    const int lane = tid & 31;
    const int row  = lane & 15;          // M (for A) / N (for B,C)
    const int hi   = lane >> 4;          // lane half selects K-subset

    const int  tile = blockIdx.x * 8 + wave;       // 16-token tile id
    const long tok0 = (long)tile * 16;
    const long trow = tok0 + row;

    // ---- Load A: 16x256 activation tile as 8 chunks of 16x32 bf16.
    // Per ISA A-layout (16-bit 16x32): lanes 0-15 hold K {0..7,16..23},
    // lanes 16-31 hold K {8..15,24..31}; element order = K ascending per run.
    v16bf A[8];
    {
        const int kbase = hi ? 8 : 0;
        #pragma unroll
        for (int kk = 0; kk < 8; ++kk) {
            const float* src = (kk < 4) ? (k + trow * DD + kk * 32)
                                        : (v + trow * DD + (kk - 4) * 32);
            const float4 f0 = *(const float4*)(src + kbase + 0);
            const float4 f1 = *(const float4*)(src + kbase + 4);
            const float4 f2 = *(const float4*)(src + kbase + 16);
            const float4 f3 = *(const float4*)(src + kbase + 20);
            v16bf a;
            a[0]  = (__bf16)f0.x; a[1]  = (__bf16)f0.y; a[2]  = (__bf16)f0.z; a[3]  = (__bf16)f0.w;
            a[4]  = (__bf16)f1.x; a[5]  = (__bf16)f1.y; a[6]  = (__bf16)f1.z; a[7]  = (__bf16)f1.w;
            a[8]  = (__bf16)f2.x; a[9]  = (__bf16)f2.y; a[10] = (__bf16)f2.z; a[11] = (__bf16)f2.w;
            a[12] = (__bf16)f3.x; a[13] = (__bf16)f3.y; a[14] = (__bf16)f3.z; a[15] = (__bf16)f3.w;
            A[kk] = a;
        }
    }

    // ---- 16 column tiles of W1; epilogue folds relu + W2 dot into C layout.
    float acc[8];
    #pragma unroll
    for (int r = 0; r < 8; ++r) acc[r] = 0.0f;

    const int fb = hi * 8;    // B layout: lanes 0-15 K={0..7,16..23}, lanes 16-31 K={8..15,24..31}
    for (int n = 0; n < 16; ++n) {
        const int ccol = n * 16 + row;     // hidden unit owned by this lane (N)
        v8f C = {0.f,0.f,0.f,0.f,0.f,0.f,0.f,0.f};
        #pragma unroll
        for (int kk = 0; kk < 8; ++kk) {
            typedef __attribute__((ext_vector_type(8))) __bf16 v8bf_t;
            const v8bf_t b0 = *(const v8bf_t*)&w1t[ccol * FPAD + kk * 32 + fb];
            const v8bf_t b1v = *(const v8bf_t*)&w1t[ccol * FPAD + kk * 32 + 16 + fb];
            const v16bf Bt = __builtin_shufflevector(b0, b1v, 0,1,2,3,4,5,6,7,8,9,10,11,12,13,14,15);
            C = __builtin_amdgcn_wmma_f32_16x16x32_bf16(
                    false, A[kk], false, Bt, (short)0, C, false, false);
        }
        const float bb = b1s[ccol];
        const float ww = w2s[ccol];
        #pragma unroll
        for (int r = 0; r < 8; ++r) {      // element r -> M = r + 8*hi
            float h = C[r] + bb;
            h = h > 0.0f ? h : 0.0f;
            acc[r] += h * ww;
        }
    }

    // Reduce over N (16 lanes per half-wave) -> per-token score
    #pragma unroll
    for (int r = 0; r < 8; ++r) {
        float x = acc[r];
        x += __shfl_xor(x, 1, 32);
        x += __shfl_xor(x, 2, 32);
        x += __shfl_xor(x, 4, 32);
        x += __shfl_xor(x, 8, 32);
        acc[r] = x;
    }
    if (row == 0) {
        const float b2v = b2[0];
        #pragma unroll
        for (int r = 0; r < 8; ++r) {
            const long t = tok0 + r + 8 * hi;
            const int  s = (int)(t & (SD - 1));
            scores[t] = acc[r] + b2v + (float)s * (1.0f / (float)(SD - 1));
        }
    }
}

// ---------------------------------------------------------------------------
// Block-wide exclusive scan over 1024 threads (Hillis-Steele in LDS)
// ---------------------------------------------------------------------------
__device__ __forceinline__ int scan_excl_1024(int val, int* buf, int tid)
{
    buf[tid] = val;
    __syncthreads();
    for (int off = 1; off < 1024; off <<= 1) {
        const int t = (tid >= off) ? buf[tid - off] : 0;
        __syncthreads();
        buf[tid] += t;
        __syncthreads();
    }
    const int inc = buf[tid];
    __syncthreads();
    return inc - val;
}

// ---------------------------------------------------------------------------
// Kernel 2: per-(b,h) top-2048 selection, indices emitted in ascending order
// with JAX top_k tie-break (lower index first among equal scores).
// One 1024-thread block per (b,h): radix-select threshold, then scan-compact.
// ---------------------------------------------------------------------------
__global__ __launch_bounds__(1024)
void topk_kernel(const float* __restrict__ scores, int* __restrict__ top_idx)
{
    __shared__ unsigned keys[SD];     // 32KB
    __shared__ int redbuf[1024];
    __shared__ int cnt;

    const int bh  = blockIdx.x;
    const int tid = threadIdx.x;
    const float* sc = scores + (long)bh * SD;

    // order-preserving u32 keys
    for (int i = tid; i < SD; i += 1024) {
        const unsigned b = __float_as_uint(sc[i]);
        keys[i] = ((int)b < 0) ? ~b : (b | 0x80000000u);
    }
    __syncthreads();

    // ---- radix select: T = 2048th-largest key
    unsigned prefix = 0;
    int kneed = KSEL;
    const int i0 = tid * 8;
    for (int bit = 31; bit >= 0; --bit) {
        if (tid == 0) cnt = 0;
        __syncthreads();
        const unsigned mask_hi = (bit == 31) ? 0u : (0xFFFFFFFFu << (bit + 1));
        int c = 0;
        #pragma unroll
        for (int j = 0; j < 8; ++j) {
            const unsigned u = keys[i0 + j];
            c += (int)(((u & mask_hi) == prefix) & ((u >> bit) & 1u));
        }
        c += __shfl_xor(c, 1, 32);
        c += __shfl_xor(c, 2, 32);
        c += __shfl_xor(c, 4, 32);
        c += __shfl_xor(c, 8, 32);
        c += __shfl_xor(c, 16, 32);
        if ((tid & 31) == 0) atomicAdd(&cnt, c);
        __syncthreads();
        const int total = cnt;
        if (total >= kneed) prefix |= (1u << bit);
        else                kneed -= total;
        __syncthreads();
    }
    const unsigned T = prefix;   // threshold key; kneed = #equals still needed

    // ---- compaction: all u > T, plus first `kneed` elements with u == T
    unsigned u[8];
    int eqc = 0;
    #pragma unroll
    for (int j = 0; j < 8; ++j) {
        u[j] = keys[i0 + j];
        eqc += (int)(u[j] == T);
    }
    const int eq_before = scan_excl_1024(eqc, redbuf, tid);

    int selc = 0, flags = 0, e = eq_before;
    #pragma unroll
    for (int j = 0; j < 8; ++j) {
        const bool isEq = (u[j] == T);
        const bool sel  = (u[j] > T) || (isEq && (e < kneed));
        e += (int)isEq;
        flags |= ((int)sel) << j;
        selc += (int)sel;
    }
    const int sel_before = scan_excl_1024(selc, redbuf, tid);

    int pos = sel_before;
    int* out = top_idx + (long)bh * KSEL;
    #pragma unroll
    for (int j = 0; j < 8; ++j) {
        if ((flags >> j) & 1) {
            out[pos] = i0 + j;   // ascending token index by construction
            ++pos;
        }
    }
}

// ---------------------------------------------------------------------------
// Kernel 3: gather selected rows; out = [k_sel | v_sel] flat
// 32 lanes per selected row, float4 per lane (128 floats/row).
// ---------------------------------------------------------------------------
__global__ __launch_bounds__(256)
void gather_kernel(const float* __restrict__ k, const float* __restrict__ v,
                   const int* __restrict__ top_idx, float* __restrict__ out)
{
    const int lane = threadIdx.x & 31;
    const int r    = blockIdx.x * 8 + (threadIdx.x >> 5);  // 0 .. BH*KSEL-1
    const int bh   = r >> 11;
    const int s    = top_idx[r];
    const long src = ((long)bh * SD + s) * DD;
    const long dst = (long)r * DD;
    const float4* ks = (const float4*)(k + src);
    const float4* vs = (const float4*)(v + src);
    float4* kd = (float4*)(out + dst);
    float4* vd = (float4*)(out + (long)BH * KSEL * DD + dst);
    kd[lane] = ks[lane];
    vd[lane] = vs[lane];
}

// ---------------------------------------------------------------------------
extern "C" void kernel_launch(void* const* d_in, const int* in_sizes, int n_in,
                              void* d_out, int out_size, void* d_ws, size_t ws_size,
                              hipStream_t stream)
{
    const float* k  = (const float*)d_in[0];
    const float* v  = (const float*)d_in[1];
    // d_in[2] = query (unused by the reference math)
    const float* W1 = (const float*)d_in[3];
    const float* b1 = (const float*)d_in[4];
    const float* W2 = (const float*)d_in[5];
    const float* b2 = (const float*)d_in[6];

    float* scores = (float*)d_ws;                                  // BH*S floats (2MB)
    int*   top_i  = (int*)((char*)d_ws + sizeof(float) * (size_t)BH * SD);

    // 1) scores: 32768 16-token tiles, 8 waves/block
    score_kernel<<<(BH * SD / 16) / 8, 256, 0, stream>>>(k, v, W1, b1, W2, b2, scores);
    // 2) top-2048 per (b,h)
    topk_kernel<<<BH, 1024, 0, stream>>>(scores, top_i);
    // 3) gather k/v rows
    gather_kernel<<<(BH * KSEL) / 8, 256, 0, stream>>>(k, v, top_i, (float*)d_out);
}
